// MiniCDDDDecoder_73821897884127
// MI455X (gfx1250) — matrix operand order repaired and checked by reading, not compiled, not run on hardware
//
#include <hip/hip_runtime.h>

typedef __attribute__((ext_vector_type(16))) __bf16 v16bf;
typedef __attribute__((ext_vector_type(8)))  __bf16 v8bf;
typedef __attribute__((ext_vector_type(8)))  float  v8f;
typedef int v4i __attribute__((__vector_size__(16)));

#define BATCH 64
#define VOC   40
#define EMB   32
#define L0    512
#define L1    1024
#define L2    2048
#define STEPS 50
#define CAT0  544   /* EMB + L0 */
#define CAT1  1536  /* L0 + L1  */
#define CAT2  3072  /* L1 + L2  */

#define GAS __attribute__((address_space(1)))
#define LAS __attribute__((address_space(3)))

// Async load-to-LDS pipeline depth (B-fragment ring slots per wave, 1KB each).
#define ADEPTH 8

#if defined(__has_builtin)
#if __has_builtin(__builtin_amdgcn_global_load_async_to_lds_b128) && \
    __has_builtin(__builtin_amdgcn_s_wait_asynccnt)
#define HAVE_ASYNC 1
#endif
#endif
#ifndef HAVE_ASYNC
#define HAVE_ASYNC 0
#endif

__device__ __forceinline__ float fsigmoid(float x) { return 1.0f / (1.0f + __expf(-x)); }

#define BF16_SHUF(lo, hi) \
    __builtin_shufflevector(lo, hi, 0,1,2,3,4,5,6,7,8,9,10,11,12,13,14,15)

// ---------------------------------------------------------------------------
// Core: one wave computes a full 64-row x 16-col output column strip:
// acc[mt] (mt=0..3) = A[mt*16..+16, :] x Bpacked[:, nt*16..+16].
// B packed per (kt,nt) fragment = 32 lanes x 32B contiguous (1KB), matching
// the CDNA5 16-bit B-matrix VGPR layout. Each fragment is read exactly once
// per step device-wide. A fragments: two 16B row-major loads per lane (hot).
// Async variant streams B through an LDS ring (ASYNCcnt) for deep pipelining.
// ---------------------------------------------------------------------------
template <bool UseAsync>
__device__ __forceinline__ void gemm_col(const __bf16* __restrict__ A, int lda,
                                         const __bf16* __restrict__ Bp, int nNT,
                                         int nKT, int nt, int lane,
                                         char* ringw, v8f acc[4])
{
    v8f z = {};
    acc[0] = z; acc[1] = z; acc[2] = z; acc[3] = z;
    const int r  = lane & 15;
    const int kb = (lane >> 4) * 8;
    const __bf16* arow  = A + (size_t)r * lda + kb;
    const size_t  bstep = (size_t)nNT * 512;                     // elems per kt
    const __bf16* bfrag = Bp + ((size_t)nt * 32 + lane) * 16;

#define CONSUME_LDS(KT, SLOT)                                                  \
    {                                                                          \
        const LAS v8bf* lp = (const LAS v8bf*)(lslot + (SLOT) * 1024);         \
        v8bf blo = lp[0];                                                      \
        v8bf bhi = lp[1];                                                      \
        v16bf b = BF16_SHUF(blo, bhi);                                         \
        const __bf16* ak = arow + (size_t)(KT) * 32;                           \
        _Pragma("unroll")                                                      \
        for (int mt = 0; mt < 4; ++mt) {                                       \
            const v8bf* ap = (const v8bf*)(ak + (size_t)mt * 16 * lda);        \
            v16bf a = BF16_SHUF(ap[0], ap[2]);                                 \
            acc[mt] = __builtin_amdgcn_wmma_f32_16x16x32_bf16(                 \
                false, a, false, b, (short)0, acc[mt], false, false);          \
        }                                                                      \
    }

#if HAVE_ASYNC
    if constexpr (UseAsync) {
        static_assert(ADEPTH == 8, "tail unroll assumes ADEPTH == 8");
        // requires nKT >= ADEPTH (all callers have nKT >= 17)
        char* lslot = ringw + lane * 32;
        // builtin signature: (global v4i*, local v4i*, imm offset, imm cpol)
#define ISSUE_ASYNC(P)                                                         \
        {                                                                      \
            GAS v4i* g = (GAS v4i*)(bfrag + (size_t)(P) * bstep);              \
            LAS v4i* l = (LAS v4i*)(lslot + ((P) & (ADEPTH - 1)) * 1024);      \
            __builtin_amdgcn_global_load_async_to_lds_b128(g, l, 0, 0);        \
            __builtin_amdgcn_global_load_async_to_lds_b128(g, l, 16, 0);       \
        }
        for (int p = 0; p < ADEPTH; ++p) ISSUE_ASYNC(p);
        const int body = nKT - ADEPTH;
        for (int kt = 0; kt < body; ++kt) {
            int slot = kt & (ADEPTH - 1);
            __builtin_amdgcn_s_wait_asynccnt(2 * (ADEPTH - 1)); // oldest pair done
            CONSUME_LDS(kt, slot);
            // slot reuse hazard: async LDS writes are unordered vs DS reads,
            // so drain this wave's DS reads before refilling the slot.
            asm volatile("s_wait_dscnt 0" ::: "memory");
            ISSUE_ASYNC(kt + ADEPTH);
        }
        // tail: asynccnt immediates must be integral constant expressions,
        // so unroll by hand with literal step indices.
#define TAIL_STEP(T)                                                           \
        __builtin_amdgcn_s_wait_asynccnt(2 * (ADEPTH - 1 - (T)));              \
        CONSUME_LDS(body + (T), (body + (T)) & (ADEPTH - 1));
        TAIL_STEP(0) TAIL_STEP(1) TAIL_STEP(2) TAIL_STEP(3)
        TAIL_STEP(4) TAIL_STEP(5) TAIL_STEP(6) TAIL_STEP(7)
#undef TAIL_STEP
#undef ISSUE_ASYNC
        return;
    }
#endif
    (void)ringw;
    // direct-load path (decoder + fallback)
    for (int kt = 0; kt < nKT; ++kt) {
        const v8bf* bp = (const v8bf*)(bfrag + (size_t)kt * bstep);
        v8bf blo = bp[0];
        v8bf bhi = bp[1];
        __builtin_prefetch((const void*)(bfrag + (size_t)(kt + 1) * bstep), 0, 3);
        v16bf b = BF16_SHUF(blo, bhi);
        const __bf16* ak = arow + (size_t)kt * 32;
#pragma unroll
        for (int mt = 0; mt < 4; ++mt) {
            const v8bf* ap = (const v8bf*)(ak + (size_t)mt * 16 * lda);
            v16bf a = BF16_SHUF(ap[0], ap[2]);
            acc[mt] = __builtin_amdgcn_wmma_f32_16x16x32_bf16(
                false, a, false, b, (short)0, acc[mt], false, false);
        }
    }
#undef CONSUME_LDS
}

// ---------------------------------------------------------------------------
// One-time: pack fp32 weight (K x N row-major) into bf16 WMMA-B fragments.
// Packed element index = ((kt*nNT + nt)*32 + lane)*16 + e ;
//   k = kt*32 + (e<8 ? (lane/16)*8 + e : 16 + (lane/16)*8 + e-8) , n = nt*16 + (lane%16). N padded to Npad.
// ---------------------------------------------------------------------------
__global__ void pack_weight_kernel(const float* __restrict__ W, __bf16* __restrict__ P,
                                   int K, int N, int Npad)
{
    size_t tid = (size_t)blockIdx.x * blockDim.x + threadIdx.x;
    size_t total = (size_t)K * Npad;
    if (tid >= total) return;
    int e    = (int)(tid & 15);
    int lane = (int)((tid >> 4) & 31);
    size_t frag = tid >> 9;
    int nNT = Npad >> 4;
    int nt = (int)(frag % nNT);
    int kt = (int)(frag / nNT);
    int k = kt * 32 + (e < 8 ? (lane >> 4) * 8 + e : 16 + (lane >> 4) * 8 + (e - 8));
    int n = nt * 16 + (lane & 15);
    float v = (n < N) ? W[(size_t)k * N + n] : 0.0f;
    P[tid] = (__bf16)v;
}

// x0 = emb[START_TOK] into cat0 x-part; done[0] = 0
__global__ void init_misc_kernel(const float* __restrict__ emb,
                                 __bf16* __restrict__ catG0, __bf16* __restrict__ catC0,
                                 int* __restrict__ done)
{
    int tid = blockIdx.x * blockDim.x + threadIdx.x;
    if (tid < BATCH * EMB) {
        int row = tid >> 5, c = tid & 31;
        __bf16 e = (__bf16)emb[39 * EMB + c];   // START_TOK = 39
        catG0[(size_t)row * CAT0 + c] = e;
        catC0[(size_t)row * CAT0 + c] = e;
    }
    if (tid < BATCH) done[tid] = 0;
}

// y = input_vecs @ dense_W + dense_b  -> h0,h1,h2 (f32) + bf16 copies in catG h-parts
__global__ void dense_init_kernel(const float* __restrict__ x, const float* __restrict__ W,
                                  const float* __restrict__ b,
                                  float* __restrict__ h0, float* __restrict__ h1, float* __restrict__ h2,
                                  __bf16* __restrict__ catG0, __bf16* __restrict__ catG1,
                                  __bf16* __restrict__ catG2)
{
    int tid = blockIdx.x * blockDim.x + threadIdx.x;
    if (tid >= BATCH * 3584) return;
    int row = tid / 3584, col = tid % 3584;
    float acc = b[col];
    const float* xr = x + row * 512;
    for (int k = 0; k < 512; ++k) acc += xr[k] * W[(size_t)k * 3584 + col];
    if (col < L0) {
        h0[row * L0 + col] = acc;
        catG0[(size_t)row * CAT0 + EMB + col] = (__bf16)acc;
    } else if (col < L0 + L1) {
        int c = col - L0;
        h1[row * L1 + c] = acc;
        catG1[(size_t)row * CAT1 + L0 + c] = (__bf16)acc;
    } else {
        int c = col - L0 - L1;
        h2[row * L2 + c] = acc;
        catG2[(size_t)row * CAT2 + L1 + c] = (__bf16)acc;
    }
}

// gates = sigmoid([x|h] @ Kg + bg) ; r-half fused into catC (r*h, bf16), z-half -> zbuf f32
__global__ void gru_gates_kernel(const __bf16* __restrict__ catG, int lda,
                                 const __bf16* __restrict__ Kgp, const float* __restrict__ bg,
                                 int nKT, int nNT, int NrT,
                                 const float* __restrict__ h, int Nout,
                                 float* __restrict__ zbuf,
                                 __bf16* __restrict__ catC, int ldc, int xw)
{
#if HAVE_ASYNC
    __shared__ __align__(16) char ring[8 * ADEPTH * 1024];
    char* ringw = ring + (size_t)(threadIdx.x >> 5) * (ADEPTH * 1024);
#else
    char* ringw = nullptr;
#endif
    int lane = threadIdx.x & 31;
    int nt = blockIdx.x * (blockDim.x >> 5) + (threadIdx.x >> 5);
    if (nt >= nNT) return;                 // wave-uniform: EXEC all-ones for WMMA
    v8f acc[4];
    gemm_col<HAVE_ASYNC != 0>(catG, lda, Kgp, nNT, nKT, nt, lane, ringw, acc);
    int n  = nt * 16 + (lane & 15);
    int mb = 8 * (lane >> 4);
    float bias = bg[n];
    if (nt < NrT) {                        // r half -> catC h-part = r*h (bf16)
#pragma unroll
        for (int mt = 0; mt < 4; ++mt)
#pragma unroll
            for (int v = 0; v < 8; ++v) {
                int m = mt * 16 + mb + v;
                float g = fsigmoid(acc[mt][v] + bias);
                catC[(size_t)m * ldc + xw + n] = (__bf16)(g * h[(size_t)m * Nout + n]);
            }
    } else {                               // z half -> zbuf (f32)
        int zc = n - NrT * 16;
#pragma unroll
        for (int mt = 0; mt < 4; ++mt)
#pragma unroll
            for (int v = 0; v < 8; ++v) {
                int m = mt * 16 + mb + v;
                zbuf[(size_t)m * Nout + zc] = fsigmoid(acc[mt][v] + bias);
            }
    }
}

// c = tanh([x|r*h] @ Kc + bc) ; n = z*h + (1-z)*c -> nf32 and bf16 into next-layer inputs
__global__ void gru_cand_kernel(const __bf16* __restrict__ catC, int lda,
                                const __bf16* __restrict__ Kcp, const float* __restrict__ bc,
                                int nKT, int nNT,
                                const float* __restrict__ zbuf, const float* __restrict__ h,
                                float* __restrict__ nf32,
                                __bf16* __restrict__ dst1, int ld1,
                                __bf16* __restrict__ dst2, int ld2)
{
#if HAVE_ASYNC
    __shared__ __align__(16) char ring[8 * ADEPTH * 1024];
    char* ringw = ring + (size_t)(threadIdx.x >> 5) * (ADEPTH * 1024);
#else
    char* ringw = nullptr;
#endif
    int lane = threadIdx.x & 31;
    int nt = blockIdx.x * (blockDim.x >> 5) + (threadIdx.x >> 5);
    if (nt >= nNT) return;
    int Nout = nNT * 16;
    v8f acc[4];
    gemm_col<HAVE_ASYNC != 0>(catC, lda, Kcp, nNT, nKT, nt, lane, ringw, acc);
    int n  = nt * 16 + (lane & 15);
    int mb = 8 * (lane >> 4);
    float bias = bc[n];
#pragma unroll
    for (int mt = 0; mt < 4; ++mt)
#pragma unroll
        for (int v = 0; v < 8; ++v) {
            int m = mt * 16 + mb + v;
            float c  = tanhf(acc[mt][v] + bias);
            float zg = zbuf[(size_t)m * Nout + n];
            float hv = h[(size_t)m * Nout + n];
            float nv = zg * hv + (1.0f - zg) * c;
            nf32[(size_t)m * Nout + n] = nv;
            __bf16 nb = (__bf16)nv;
            dst1[(size_t)m * ld1 + n] = nb;
            if (dst2) dst2[(size_t)m * ld2 + n] = nb;
        }
}

// Single-WG decoder: logits = n2 @ dec_W (N padded 40->48), argmax, masking, token & x update.
__global__ void decoder_kernel(const __bf16* __restrict__ decIn, const __bf16* __restrict__ decp,
                               const float* __restrict__ emb,
                               int step, int* __restrict__ done,
                               __bf16* __restrict__ catG0, __bf16* __restrict__ catC0,
                               int* __restrict__ outToks, float* __restrict__ outLogits)
{
    __shared__ float slog[BATCH * 48];
    int lane = threadIdx.x & 31;
    int w    = threadIdx.x >> 5;
    if (w < 3) {                           // 3 N-tiles, each wave does all 4 M-tiles
        v8f acc[4];
        gemm_col<false>(decIn, L2, decp, 3, L2 / 32, w, lane, nullptr, acc);
        int n  = w * 16 + (lane & 15);
        int mb = 8 * (lane >> 4);
#pragma unroll
        for (int mt = 0; mt < 4; ++mt)
#pragma unroll
            for (int v = 0; v < 8; ++v)
                slog[(mt * 16 + mb + v) * 48 + n] = acc[mt][v];
    }
    __syncthreads();
    int row = threadIdx.x;
    if (row < BATCH) {
        int dprev = done[step * BATCH + row];
        float best = slog[row * 48 + 0];
        int bi = 0;
        for (int c = 1; c < VOC; ++c) {
            float vl = slog[row * 48 + c];
            if (vl > best) { best = vl; bi = c; }   // first-max tie break (jnp.argmax)
        }
        int tok = dprev ? 0 : bi;                   // STOP_TOK = 0
        outToks[row * STEPS + step] = tok;
        float* lp = outLogits + ((size_t)row * STEPS + step) * VOC;
        for (int c = 0; c < VOC; ++c) lp[c] = dprev ? 0.0f : slog[row * 48 + c];
        done[(step + 1) * BATCH + row] = dprev | (bi == 0);
        if (!dprev) {                               // x = emb[idx]
            for (int c = 0; c < EMB; ++c) {
                __bf16 e = (__bf16)emb[bi * EMB + c];
                catG0[(size_t)row * CAT0 + c] = e;
                catC0[(size_t)row * CAT0 + c] = e;
            }
        }
    }
}

// h_i = done ? h_i : n_i  (uses pre-step done) ; also refresh bf16 catG h-parts
__global__ void h_update_kernel(int step, const int* __restrict__ done,
                                const float* __restrict__ n0, float* __restrict__ h0, __bf16* __restrict__ catG0,
                                const float* __restrict__ n1, float* __restrict__ h1, __bf16* __restrict__ catG1,
                                const float* __restrict__ n2, float* __restrict__ h2, __bf16* __restrict__ catG2)
{
    int tid = blockIdx.x * blockDim.x + threadIdx.x;
    if (tid >= BATCH * 3584) return;
    int row = tid / 3584, col = tid % 3584;
    int d = done[step * BATCH + row];
    if (col < L0) {
        float nh = d ? h0[row * L0 + col] : n0[row * L0 + col];
        h0[row * L0 + col] = nh;
        catG0[(size_t)row * CAT0 + EMB + col] = (__bf16)nh;
    } else if (col < L0 + L1) {
        int c = col - L0;
        float nh = d ? h1[row * L1 + c] : n1[row * L1 + c];
        h1[row * L1 + c] = nh;
        catG1[(size_t)row * CAT1 + L0 + c] = (__bf16)nh;
    } else {
        int c = col - L0 - L1;
        float nh = d ? h2[row * L2 + c] : n2[row * L2 + c];
        h2[row * L2 + c] = nh;
        catG2[(size_t)row * CAT2 + L1 + c] = (__bf16)nh;
    }
}

extern "C" void kernel_launch(void* const* d_in, const int* in_sizes, int n_in,
                              void* d_out, int out_size, void* d_ws, size_t ws_size,
                              hipStream_t stream)
{
    (void)in_sizes; (void)n_in; (void)out_size; (void)ws_size;
    const float* input_vecs = (const float*)d_in[0];
    const float* emb        = (const float*)d_in[1];
    const float* dense_W    = (const float*)d_in[2];
    const float* dense_b    = (const float*)d_in[3];
    const float* dec_W      = (const float*)d_in[4];
    const float* Kg0 = (const float*)d_in[5];  const float* bg0 = (const float*)d_in[6];
    const float* Kc0 = (const float*)d_in[7];  const float* bc0 = (const float*)d_in[8];
    const float* Kg1 = (const float*)d_in[9];  const float* bg1 = (const float*)d_in[10];
    const float* Kc1 = (const float*)d_in[11]; const float* bc1 = (const float*)d_in[12];
    const float* Kg2 = (const float*)d_in[13]; const float* bg2 = (const float*)d_in[14];
    const float* Kc2 = (const float*)d_in[15]; const float* bc2 = (const float*)d_in[16];

    char* ws = (char*)d_ws;
    size_t off = 0;
    auto alloc = [&](size_t bytes) -> char* {
        char* p = ws + off;
        off += (bytes + 255) & ~(size_t)255;
        return p;
    };

    // packed bf16 weights (~49 MB, resident in the 192 MB L2 across all 50 steps)
    __bf16* Kg0p = (__bf16*)alloc((size_t)CAT0 * 1024 * 2);
    __bf16* Kc0p = (__bf16*)alloc((size_t)CAT0 * 512  * 2);
    __bf16* Kg1p = (__bf16*)alloc((size_t)CAT1 * 2048 * 2);
    __bf16* Kc1p = (__bf16*)alloc((size_t)CAT1 * 1024 * 2);
    __bf16* Kg2p = (__bf16*)alloc((size_t)CAT2 * 4096 * 2);
    __bf16* Kc2p = (__bf16*)alloc((size_t)CAT2 * 2048 * 2);
    __bf16* decp = (__bf16*)alloc((size_t)L2   * 48   * 2);
    // activations
    __bf16* catG0 = (__bf16*)alloc((size_t)BATCH * CAT0 * 2);
    __bf16* catC0 = (__bf16*)alloc((size_t)BATCH * CAT0 * 2);
    __bf16* catG1 = (__bf16*)alloc((size_t)BATCH * CAT1 * 2);
    __bf16* catC1 = (__bf16*)alloc((size_t)BATCH * CAT1 * 2);
    __bf16* catG2 = (__bf16*)alloc((size_t)BATCH * CAT2 * 2);
    __bf16* catC2 = (__bf16*)alloc((size_t)BATCH * CAT2 * 2);
    __bf16* decIn = (__bf16*)alloc((size_t)BATCH * L2   * 2);
    float* h0 = (float*)alloc((size_t)BATCH * L0 * 4);
    float* h1 = (float*)alloc((size_t)BATCH * L1 * 4);
    float* h2 = (float*)alloc((size_t)BATCH * L2 * 4);
    float* n0 = (float*)alloc((size_t)BATCH * L0 * 4);
    float* n1 = (float*)alloc((size_t)BATCH * L1 * 4);
    float* n2 = (float*)alloc((size_t)BATCH * L2 * 4);
    float* z0 = (float*)alloc((size_t)BATCH * L0 * 4);
    float* z1 = (float*)alloc((size_t)BATCH * L1 * 4);
    float* z2 = (float*)alloc((size_t)BATCH * L2 * 4);
    int*  done = (int*)alloc((size_t)(STEPS + 1) * BATCH * 4);

    int*   outToks   = (int*)d_out;                        // (64,50) int32 bits
    float* outLogits = (float*)d_out + BATCH * STEPS;      // (64,50,40) f32

    auto cdiv = [](int a, int b) { return (a + b - 1) / b; };
    const int TB = 256;   // 8 waves

    // ---- one-time per call: pack weights, init state -----------------------
    pack_weight_kernel<<<cdiv(CAT0 * 1024, TB), TB, 0, stream>>>(Kg0, Kg0p, CAT0, 1024, 1024);
    pack_weight_kernel<<<cdiv(CAT0 * 512 , TB), TB, 0, stream>>>(Kc0, Kc0p, CAT0, 512 , 512 );
    pack_weight_kernel<<<cdiv(CAT1 * 2048, TB), TB, 0, stream>>>(Kg1, Kg1p, CAT1, 2048, 2048);
    pack_weight_kernel<<<cdiv(CAT1 * 1024, TB), TB, 0, stream>>>(Kc1, Kc1p, CAT1, 1024, 1024);
    pack_weight_kernel<<<cdiv(CAT2 * 4096, TB), TB, 0, stream>>>(Kg2, Kg2p, CAT2, 4096, 4096);
    pack_weight_kernel<<<cdiv(CAT2 * 2048, TB), TB, 0, stream>>>(Kc2, Kc2p, CAT2, 2048, 2048);
    pack_weight_kernel<<<cdiv(L2 * 48   , TB), TB, 0, stream>>>(dec_W, decp, L2, VOC, 48);
    init_misc_kernel<<<cdiv(BATCH * EMB, TB), TB, 0, stream>>>(emb, catG0, catC0, done);
    dense_init_kernel<<<cdiv(BATCH * 3584, TB), TB, 0, stream>>>(
        input_vecs, dense_W, dense_b, h0, h1, h2, catG0, catG1, catG2);

    // ---- 50 autoregressive steps (one wave per 16-col output strip) --------
    for (int s = 0; s < STEPS; ++s) {
        // layer 0: K=544 (17 kt)
        gru_gates_kernel<<<cdiv(64, 8), TB, 0, stream>>>(
            catG0, CAT0, Kg0p, bg0, 17, 64, 32, h0, L0, z0, catC0, CAT0, EMB);
        gru_cand_kernel<<<cdiv(32, 8), TB, 0, stream>>>(
            catC0, CAT0, Kc0p, bc0, 17, 32, z0, h0, n0, catG1, CAT1, catC1, CAT1);
        // layer 1: K=1536 (48 kt)
        gru_gates_kernel<<<cdiv(128, 8), TB, 0, stream>>>(
            catG1, CAT1, Kg1p, bg1, 48, 128, 64, h1, L1, z1, catC1, CAT1, L0);
        gru_cand_kernel<<<cdiv(64, 8), TB, 0, stream>>>(
            catC1, CAT1, Kc1p, bc1, 48, 64, z1, h1, n1, catG2, CAT2, catC2, CAT2);
        // layer 2: K=3072 (96 kt)
        gru_gates_kernel<<<cdiv(256, 8), TB, 0, stream>>>(
            catG2, CAT2, Kg2p, bg2, 96, 256, 128, h2, L2, z2, catC2, CAT2, L1);
        gru_cand_kernel<<<cdiv(128, 8), TB, 0, stream>>>(
            catC2, CAT2, Kc2p, bc2, 96, 128, z2, h2, n2, decIn, L2, (__bf16*)nullptr, 0);
        // decoder + argmax + token/done/x updates
        decoder_kernel<<<1, TB, 0, stream>>>(
            decIn, decp, emb, s, done, catG0, catC0, outToks, outLogits);
        // masked recurrent state update (uses pre-step done)
        h_update_kernel<<<cdiv(BATCH * 3584, TB), TB, 0, stream>>>(
            s, done, n0, h0, catG0, n1, h1, catG1, n2, h2, catG2);
    }
}
